// PerspectiveConv2d_40432822124906
// MI455X (gfx1250) — matrix operand order repaired and checked by reading, not compiled, not run on hardware
//
#include <hip/hip_runtime.h>

// ---------------- problem constants ----------------
#define B_N    4
#define C_IN   128
#define H_IN   96
#define W_IN   320
#define COUT   128
#define KK     9                    // 3x3
#define KDIM   (KK * C_IN)          // 1152 (im2col rows, ordering k*128+c)
#define NPIX   (H_IN * W_IN)        // 30720 per batch
#define TILE_N 32                   // output pixels per block
#define TILES_PER_B (NPIX / TILE_N) // 960
#define NPAIR  (TILE_N * KK)        // 288 (pixel,k) pairs per block

typedef __attribute__((ext_vector_type(16))) __bf16 v16bf;
typedef __attribute__((ext_vector_type(8)))  __bf16 v8bf;
typedef __attribute__((ext_vector_type(8)))  float  v8f;

union FragA { v16bf v; v8bf h[2]; };

__device__ __forceinline__ unsigned short f2bf(float f) {
    union { float f; unsigned u; } x; x.f = f;
    unsigned r = x.u + 0x7FFFu + ((x.u >> 16) & 1u);  // round-to-nearest-even
    return (unsigned short)(r >> 16);
}

// ---------------- kernel 0: NCHW -> NHWC ----------------
__global__ __launch_bounds__(256)
void nchw_to_nhwc(const float* __restrict__ in, float* __restrict__ out) {
    __shared__ float tile[32][33];
    const int bz = blockIdx.z;          // b*H + y
    const int bb = bz / H_IN;
    const int y  = bz % H_IN;
    const int x0 = blockIdx.x * 32;
    const int cT = blockIdx.y * 32;
    const int tx = threadIdx.x;         // 32
    const int ty = threadIdx.y;         // 8
#pragma unroll
    for (int i = 0; i < 4; ++i) {
        int c = cT + ty + 8 * i;
        tile[ty + 8 * i][tx] =
            in[(((size_t)bb * C_IN + c) * H_IN + y) * W_IN + x0 + tx];
    }
    __syncthreads();
#pragma unroll
    for (int i = 0; i < 4; ++i) {
        int xx = ty + 8 * i;            // x within tile
        out[(((size_t)bb * H_IN + y) * W_IN + (x0 + xx)) * C_IN + cT + tx] =
            tile[tx][xx];
    }
}

// -------- kernel 1: weight fp32 -> bf16, chunk-major [kd/32][o][kd%32] --------
__global__ __launch_bounds__(256)
void weight_to_bf16(const float* __restrict__ w, unsigned short* __restrict__ wb) {
    int tid    = blockIdx.x * 256 + threadIdx.x;   // exactly 128*1152 threads
    int within = tid & 31;                          // kd % 32
    int o      = (tid >> 5) & 127;                  // Cout row
    int chunk  = tid >> 12;                         // kd / 32  (0..35)
    int kd     = chunk * 32 + within;               // im2col column = k*128+c
    int k      = kd >> 7;
    int c      = kd & 127;
    float f = w[((size_t)(o * C_IN + c)) * KK + k];
    wb[tid] = f2bf(f);                              // coalesced bf16 store
}

// ---------------- kernel 2: fused deformable sampling + WMMA GEMM ----------------
__global__ __launch_bounds__(256)
void deform_conv_wmma(const float* __restrict__ feat_nhwc,
                      const float* __restrict__ offsets,
                      const unsigned short* __restrict__ wb,
                      float* __restrict__ out) {
    // im2col B tile: [pixel][k*128+c] bf16 == [pr*128+c], 32x1152 -> 73728 B
    __shared__ __align__(32) unsigned short Bs[TILE_N * KDIM];
    // per-(pixel,k) bilinear records: weights + 4 tap element-offsets (-1 = OOB)
    __shared__ float4 Wt[NPAIR];   // 4608 B
    __shared__ int4   Of[NPAIR];   // 4608 B

    const int block = blockIdx.x;
    const int b     = block / TILES_PER_B;
    const int tile  = block % TILES_PER_B;
    const int pix0  = tile * TILE_N;
    const int ho    = pix0 / W_IN;          // whole tile shares ho (320 % 32 == 0)
    const int wo0   = pix0 % W_IN;

    const int tid  = threadIdx.x;
    const int wave = tid >> 5;
    const int lane = tid & 31;

    const float* fb = feat_nhwc + (size_t)b * H_IN * W_IN * C_IN;
    const float* ob = offsets   + (size_t)b * (2 * KK) * NPIX + (size_t)ho * W_IN;

    // ---------------- phase 0: per-pair bilinear precompute ----------------
    for (int pr = tid; pr < NPAIR; pr += 256) {
        int p  = pr / KK;
        int k  = pr % KK;
        int ky = k / 3, kx = k % 3;
        int wo = wo0 + p;

        float offy = ob[(size_t)(2 * k)     * NPIX + wo];
        float offx = ob[(size_t)(2 * k + 1) * NPIX + wo];
        float py = (float)(ho - 1 + ky) + offy;
        float px = (float)(wo - 1 + kx) + offx;

        float y0f = floorf(py), x0f = floorf(px);
        float wy1 = py - y0f,   wx1 = px - x0f;
        float wy0 = 1.0f - wy1, wx0 = 1.0f - wx1;
        int y0 = (int)y0f, x0 = (int)x0f;
        int y1 = y0 + 1,   x1 = x0 + 1;
        bool vy0 = (y0 >= 0) & (y0 < H_IN);
        bool vy1 = (y1 >= 0) & (y1 < H_IN);
        bool vx0 = (x0 >= 0) & (x0 < W_IN);
        bool vx1 = (x1 >= 0) & (x1 < W_IN);

        Wt[pr] = make_float4(wy0 * wx0, wy0 * wx1, wy1 * wx0, wy1 * wx1);
        Of[pr] = make_int4((vy0 & vx0) ? (y0 * W_IN + x0) * C_IN : -1,
                           (vy0 & vx1) ? (y0 * W_IN + x1) * C_IN : -1,
                           (vy1 & vx0) ? (y1 * W_IN + x0) * C_IN : -1,
                           (vy1 & vx1) ? (y1 * W_IN + x1) * C_IN : -1);
    }
    __syncthreads();

    // ---------------- phase 1: sampling into LDS B tile ----------------
    const int c0 = lane * 4;                // 4 channels per lane

#pragma unroll 4
    for (int i = 0; i < NPAIR / 8; ++i) {   // 36 pairs per wave
        int pr = wave * (NPAIR / 8) + i;
        float4 wv = Wt[pr];
        int4   ov = Of[pr];

        float a0 = 0.f, a1 = 0.f, a2 = 0.f, a3 = 0.f;
        if (ov.x >= 0) {
            const float4 t = *(const float4*)(fb + ov.x + c0);
            a0 += t.x * wv.x; a1 += t.y * wv.x; a2 += t.z * wv.x; a3 += t.w * wv.x;
        }
        if (ov.y >= 0) {
            const float4 t = *(const float4*)(fb + ov.y + c0);
            a0 += t.x * wv.y; a1 += t.y * wv.y; a2 += t.z * wv.y; a3 += t.w * wv.y;
        }
        if (ov.z >= 0) {
            const float4 t = *(const float4*)(fb + ov.z + c0);
            a0 += t.x * wv.z; a1 += t.y * wv.z; a2 += t.z * wv.z; a3 += t.w * wv.z;
        }
        if (ov.w >= 0) {
            const float4 t = *(const float4*)(fb + ov.w + c0);
            a0 += t.x * wv.w; a1 += t.y * wv.w; a2 += t.z * wv.w; a3 += t.w * wv.w;
        }

        unsigned long long pack =
              (unsigned long long)f2bf(a0)
            | ((unsigned long long)f2bf(a1) << 16)
            | ((unsigned long long)f2bf(a2) << 32)
            | ((unsigned long long)f2bf(a3) << 48);
        // B-tile address: pr*128 + c0  (since row ordering is k*128+c, pr = p*9+k)
        *(unsigned long long*)&Bs[pr * C_IN + c0] = pack;   // ds_store_b64
    }

    __syncthreads();

    // ---------------- phase 2: WMMA GEMM ----------------
    // Wave w owns Cout rows [16w,16w+16), N tiles {0,1} (pixels 0-15, 16-31).
    const int half = lane >> 4;
    const int idx  = lane & 15;
    const int row  = wave * 16 + idx;                    // A row (Cout)

    v8f acc0 = {};
    v8f acc1 = {};

#pragma unroll 4
    for (int kb = 0; kb < KDIM; kb += 32) {
        // A chunk layout: wb[(kb/32)*COUT*32 + row*32 + kwithin] -> dense 8 KB chunk
        // A (16x32 bf16): half 0 -> K {0..15}, half 1 -> K {16..31}
        const unsigned short* wbc = wb + (size_t)(kb >> 5) * (COUT * 32) + row * 32;
        FragA A;
        A.h[0] = *(const v8bf*)(wbc + half * 16);        // 16B global
        A.h[1] = *(const v8bf*)(wbc + half * 16 + 8);    // 16B global

        // B (32x16 bf16): lanes 0-15 K = kb..kb+15, lanes 16-31 K = kb+16..kb+31
        v16bf B0 = *(const v16bf*)(&Bs[(idx)      * KDIM + kb + half * 16]);
        v16bf B1 = *(const v16bf*)(&Bs[(16 + idx) * KDIM + kb + half * 16]);

        acc0 = __builtin_amdgcn_wmma_f32_16x16x32_bf16(
                   false, A.v, false, B0, (short)0, acc0, false, false);
        acc1 = __builtin_amdgcn_wmma_f32_16x16x32_bf16(
                   false, A.v, false, B1, (short)0, acc1, false, false);
    }

    // ---- store (C/D layout: lanes<16 -> M=v, lanes>=16 -> M=8+v; N=idx) ----
    float* outb = out + (size_t)b * COUT * NPIX;
#pragma unroll
    for (int v = 0; v < 8; ++v) {
        int o = wave * 16 + half * 8 + v;
        size_t base = (size_t)o * NPIX + pix0 + idx;
        outb[base]      = acc0[v];
        outb[base + 16] = acc1[v];
    }
}

// ---------------- launch ----------------
extern "C" void kernel_launch(void* const* d_in, const int* in_sizes, int n_in,
                              void* d_out, int out_size, void* d_ws, size_t ws_size,
                              hipStream_t stream) {
    const float* features = (const float*)d_in[0];   // [4,128,96,320]
    const float* offsets  = (const float*)d_in[1];   // [4,18,96,320]
    const float* weight   = (const float*)d_in[2];   // [128,128,3,3]
    float* out = (float*)d_out;                      // [4,128,96,320]

    // workspace: NHWC features (62.9 MB) + bf16 weight (288 KB)
    float* feat_nhwc = (float*)d_ws;
    const size_t feat_bytes = (size_t)B_N * H_IN * W_IN * C_IN * sizeof(float);
    unsigned short* wb = (unsigned short*)((char*)d_ws + feat_bytes);

    nchw_to_nhwc<<<dim3(W_IN / 32, C_IN / 32, B_N * H_IN), dim3(32, 8, 1), 0, stream>>>(
        features, feat_nhwc);

    weight_to_bf16<<<(COUT * KDIM) / 256, 256, 0, stream>>>(weight, wb);

    deform_conv_wmma<<<B_N * TILES_PER_B, 256, 0, stream>>>(
        feat_nhwc, offsets, wb, out);
}
